// Transformer_87608742904081
// MI455X (gfx1250) — matrix operand order repaired and checked by reading, not compiled, not run on hardware
//
#include <hip/hip_runtime.h>
#include <hip/hip_bf16.h>
#include <math.h>

// Problem dims (compile-time)
#define BB   16
#define TT   1023
#define CCH  384
#define HH   6
#define DHH  64
#define LLN  6
#define VV   256
#define FFF  (4*CCH)        // 1536
#define MMR  (BB*TT)        // 16368 rows, multiple of 16
#define EPSV 1e-5f

typedef __attribute__((ext_vector_type(16))) _Float16 v16h;
typedef __attribute__((ext_vector_type(8)))  _Float16 v8h;
typedef __attribute__((ext_vector_type(8)))  float    v8f;

union Pack32 { _Float16 h[32]; uint4 u[4]; };

__device__ __forceinline__ v8f wmma_f16(v16h a, v16h b, v8f c) {
    // D = A(16x32 f16) * B(32x16 f16) + C(16x16 f32)
    return __builtin_amdgcn_wmma_f32_16x16x32_f16(
        /*neg_a=*/false, a, /*neg_b=*/false, b,
        /*c_mod=*/(short)0, c, /*reuse_a=*/false, /*reuse_b=*/false);
}

// Build A-layout fragment from a row of LDS: halves {base+hi*16..+7, base+hi*16+8..+15}
__device__ __forceinline__ v16h frag_a_row(const _Float16* row, int base, int hi) {
    const v8h lo = *(const v8h*)(row + base + hi * 16);
    const v8h hh = *(const v8h*)(row + base + hi * 16 + 8);
    return __builtin_shufflevector(lo, hh, 0,1,2,3,4,5,6,7,8,9,10,11,12,13,14,15);
}

// ---------------------------------------------------------------- fp32 -> f16
__global__ void cvt_f16_kernel(const float* __restrict__ in, _Float16* __restrict__ out, int n) {
    for (int i = blockIdx.x * blockDim.x + threadIdx.x; i < n; i += gridDim.x * blockDim.x)
        out[i] = (_Float16)in[i];
}

// fp32 [L,K,N] -> f16 transposed [L,N,K]
__global__ void cvtT_f16_kernel(const float* __restrict__ in, _Float16* __restrict__ out,
                                int K, int N, int L) {
    const int total = K * N;
    const long long tot = (long long)total * L;
    for (long long i = blockIdx.x * blockDim.x + threadIdx.x; i < tot;
         i += (long long)gridDim.x * blockDim.x) {
        const int l   = (int)(i / total);
        const int rem = (int)(i % total);
        const int r   = rem / N;
        const int c   = rem % N;
        out[(size_t)l * total + (size_t)c * K + r] = (_Float16)in[i];
    }
}

// ---------------------------------------------------------------- embedding
__global__ void embed_kernel(const int* __restrict__ x, const float* __restrict__ tok,
                             const float* __restrict__ pos, float* __restrict__ h) {
    const int bt  = blockIdx.x;          // 0..MMR-1
    const int t   = bt % TT;
    const int id  = x[bt];
    float* hr = h + (size_t)bt * CCH;
    const float* tr = tok + (size_t)id * CCH;
    const float* pr = pos + (size_t)t  * CCH;
    for (int c = threadIdx.x; c < CCH; c += blockDim.x)
        hr[c] = tr[c] + pr[c];
}

// ---------------------------------------------------------------- layernorm
__global__ void ln_kernel(const float* __restrict__ x, const float* __restrict__ g,
                          const float* __restrict__ b, float* __restrict__ y) {
    __shared__ float red[128];
    const int row = blockIdx.x;
    const int tid = threadIdx.x;
    const float* xr = x + (size_t)row * CCH;
    float s = 0.f;
    for (int c = tid; c < CCH; c += 128) s += xr[c];
    red[tid] = s; __syncthreads();
    for (int o = 64; o > 0; o >>= 1) { if (tid < o) red[tid] += red[tid + o]; __syncthreads(); }
    const float mu = red[0] * (1.0f / CCH);
    __syncthreads();
    float v = 0.f;
    for (int c = tid; c < CCH; c += 128) { float d = xr[c] - mu; v += d * d; }
    red[tid] = v; __syncthreads();
    for (int o = 64; o > 0; o >>= 1) { if (tid < o) red[tid] += red[tid + o]; __syncthreads(); }
    const float inv = rsqrtf(red[0] * (1.0f / CCH) + EPSV);
    float* yr = y + (size_t)row * CCH;
    for (int c = tid; c < CCH; c += 128)
        yr[c] = (xr[c] - mu) * inv * g[c] + b[c];
}

// ---------------------------------------------------------------- WMMA GEMM
// Y[M,N] = X[M,K] (fp32, cast to f16) @ W (given TRANSPOSED as WT[N,K], f16)
// [+bias] [+residual] [relu]. Block: 128 threads = 4 waves, tile 16(M) x 64(N),
// wave w owns cols w*16..+15. K staged 128/barrier -> 4 WMMAs per wave.
// Requires K % 128 == 0 (384, 1536 here).
__global__ void gemm_wmma_kernel(const float* __restrict__ X, const _Float16* __restrict__ WT,
                                 const float* __restrict__ bias, const float* __restrict__ res,
                                 float* __restrict__ Y, int M, int N, int K, int relu) {
    const int mt   = blockIdx.x;
    const int nt   = blockIdx.y;
    const int tid  = threadIdx.x;
    const int wave = tid >> 5;
    const int lane = tid & 31;
    const int m    = lane & 15;
    const int hi   = lane >> 4;
    const int m0   = mt * 16;
    const int n0   = nt * 64;

    __shared__ __align__(32) _Float16 As [16][128];
    __shared__ __align__(32) _Float16 BsT[64][128];   // [n][k]

    const int ar  = tid >> 3;          // 0..15  : A staging row
    const int ac0 = (tid & 7) * 16;    // 0..112 : A staging col base
    const int br  = tid >> 1;          // 0..63  : B staging row (n)
    const int bc0 = (tid & 1) * 64;    // 0/64   : B staging col base (k)

    v8f acc = {};
    for (int k0 = 0; k0 < K; k0 += 128) {
        // L2-feeding prefetch of next K stage
        if (k0 + 128 < K) {
            __builtin_prefetch(&X [(size_t)(m0 + ar) * K + (k0 + 128) + ac0], 0, 1);
            __builtin_prefetch(&WT[(size_t)(n0 + br) * K + (k0 + 128) + bc0], 0, 1);
        }
        // Stage A: 16x128 fp32 -> f16 (float4 loads, packed b128 LDS stores)
        {
            const float4* src = (const float4*)&X[(size_t)(m0 + ar) * K + k0 + ac0];
            Pack32 pk;
#pragma unroll
            for (int u = 0; u < 4; ++u) {
                const float4 f = src[u];
                pk.h[u * 4 + 0] = (_Float16)f.x;
                pk.h[u * 4 + 1] = (_Float16)f.y;
                pk.h[u * 4 + 2] = (_Float16)f.z;
                pk.h[u * 4 + 3] = (_Float16)f.w;
            }
            uint4* dst = (uint4*)&As[ar][ac0];
            dst[0] = pk.u[0];
            dst[1] = pk.u[1];
        }
        // Stage B: 64 rows (n) x 128 cols (k); 64 halves (128B) per thread
        {
            const uint4* src = (const uint4*)&WT[(size_t)(n0 + br) * K + k0 + bc0];
            uint4* dst = (uint4*)&BsT[br][bc0];
#pragma unroll
            for (int u = 0; u < 8; ++u) dst[u] = src[u];
        }
        __syncthreads();
#pragma unroll
        for (int kc = 0; kc < 4; ++kc) {
            const v16h a = frag_a_row(&As[m][0], kc * 32, hi);
            const v16h b = *(const v16h*)&BsT[wave * 16 + m][kc * 32 + hi * 16];
            acc = wmma_f16(a, b, acc);
        }
        __syncthreads();
    }
    const int n = n0 + wave * 16 + m;
#pragma unroll
    for (int r = 0; r < 8; ++r) {
        const int mr = m0 + r + hi * 8;            // C/D layout row
        float v = acc[r];
        if (bias) v += bias[n];
        if (res)  v += res[(size_t)mr * N + n];
        if (relu) v = fmaxf(v, 0.f);
        Y[(size_t)mr * N + n] = v;
    }
}

// ---------------------------------------------------------------- attention
// One wave per (query-tile, head, batch). Flash-style online softmax, WMMA for
// scores (K=64) and P@V. V is staged transposed (VtT[dim][key]) and P widened
// to 32 zero-padded key columns so every fragment is a contiguous LDS load.
__global__ void attn_kernel(const float* __restrict__ q, const float* __restrict__ k,
                            const float* __restrict__ v, float* __restrict__ o) {
    const int qt   = blockIdx.x;
    const int hh   = blockIdx.y;
    const int bb   = blockIdx.z;
    const int lane = threadIdx.x;
    const int m    = lane & 15;
    const int hi   = lane >> 4;
    const int t0   = qt * 16;

    __shared__ __align__(32) _Float16 Qt [16][64];
    __shared__ __align__(32) _Float16 Kt [16][64];
    __shared__ __align__(32) _Float16 VtT[64][32];   // [dim][key], keys 16..31 = 0
    __shared__ __align__(32) _Float16 P32[16][32];   // cols 16..31 = 0

    const size_t bh = (size_t)bb * TT * CCH + (size_t)hh * DHH;

    const int sr  = lane >> 1;          // staging row 0..15
    const int sc0 = (lane & 1) * 32;    // staging col base 0 / 32

    // Zero the static halves once (never rewritten below)
    {
        const uint4 z4 = {0, 0, 0, 0};
        for (int i = lane; i < 64 * 2; i += 32)          // VtT[d][16..31]
            ((uint4*)&VtT[i >> 1][16])[i & 1] = z4;
        ((uint4*)&P32[lane >> 1][16])[lane & 1] = z4;    // P32[r][16..31]
    }

    // Stage Q (valid-row clamp + zero-fill), packed b128 stores
    {
        const int t  = t0 + sr;
        const int tc = (t < TT) ? t : (TT - 1);
        const bool ok = (t < TT);
        const float4* src = (const float4*)&q[bh + (size_t)tc * CCH + sc0];
        Pack32 pk;
#pragma unroll
        for (int u = 0; u < 8; ++u) {
            const float4 f = src[u];
            pk.h[u * 4 + 0] = (_Float16)(ok ? f.x : 0.f);
            pk.h[u * 4 + 1] = (_Float16)(ok ? f.y : 0.f);
            pk.h[u * 4 + 2] = (_Float16)(ok ? f.z : 0.f);
            pk.h[u * 4 + 3] = (_Float16)(ok ? f.w : 0.f);
        }
        uint4* dst = (uint4*)&Qt[sr][sc0];
#pragma unroll
        for (int u = 0; u < 4; ++u) dst[u] = pk.u[u];
    }
    __syncthreads();

    const v16h aq0 = frag_a_row(&Qt[m][0], 0,  hi);
    const v16h aq1 = frag_a_row(&Qt[m][0], 32, hi);

    float rm[8], rl[8];
    v8f oacc[4] = {};
#pragma unroll
    for (int r = 0; r < 8; ++r) { rm[r] = -INFINITY; rl[r] = 0.f; }

    for (int st = 0; st <= qt; ++st) {
        __syncthreads();
        const int s0 = st * 16;
        // Prefetch next key tile while computing this one
        if (st < qt) {
            const int pn = s0 + 16 + sr;
            const int pc = (pn < TT) ? pn : (TT - 1);
            __builtin_prefetch(&k[bh + (size_t)pc * CCH + sc0], 0, 1);
            __builtin_prefetch(&v[bh + (size_t)pc * CCH + sc0], 0, 1);
        }
        // Stage K (row-major, packed) and V (transposed [dim][key])
        {
            const int s  = s0 + sr;
            const int sc = (s < TT) ? s : (TT - 1);
            const bool ok = (s < TT);
            const float4* ksrc = (const float4*)&k[bh + (size_t)sc * CCH + sc0];
            const float4* vsrc = (const float4*)&v[bh + (size_t)sc * CCH + sc0];
            Pack32 pk;
#pragma unroll
            for (int u = 0; u < 8; ++u) {
                const float4 kf = ksrc[u];
                const float4 vf = vsrc[u];
                pk.h[u * 4 + 0] = (_Float16)(ok ? kf.x : 0.f);
                pk.h[u * 4 + 1] = (_Float16)(ok ? kf.y : 0.f);
                pk.h[u * 4 + 2] = (_Float16)(ok ? kf.z : 0.f);
                pk.h[u * 4 + 3] = (_Float16)(ok ? kf.w : 0.f);
                VtT[sc0 + u * 4 + 0][sr] = (_Float16)(ok ? vf.x : 0.f);
                VtT[sc0 + u * 4 + 1][sr] = (_Float16)(ok ? vf.y : 0.f);
                VtT[sc0 + u * 4 + 2][sr] = (_Float16)(ok ? vf.z : 0.f);
                VtT[sc0 + u * 4 + 3][sr] = (_Float16)(ok ? vf.w : 0.f);
            }
            uint4* dst = (uint4*)&Kt[sr][sc0];
#pragma unroll
            for (int u = 0; u < 4; ++u) dst[u] = pk.u[u];
        }
        __syncthreads();

        // Scores: contiguous B fragments from Kt rows
        const v16h bk0 = *(const v16h*)&Kt[m][hi * 16];
        const v16h bk1 = *(const v16h*)&Kt[m][32 + hi * 16];
        v8f sacc = {};
        sacc = wmma_f16(aq0, bk0, sacc);
        sacc = wmma_f16(aq1, bk1, sacc);

        const int s = s0 + m;             // this lane's key column
        float pv[8];
#pragma unroll
        for (int r = 0; r < 8; ++r) {
            const int t = t0 + r + hi * 8;
            float val = sacc[r] * 0.125f;                 // DH^-0.5
            if (s > t || val == 0.f) val = -INFINITY;     // faithful masking quirk
            pv[r] = val;
        }
#pragma unroll
        for (int r = 0; r < 8; ++r) {
            float x = pv[r];
            for (int msk = 1; msk < 16; msk <<= 1) x = fmaxf(x, __shfl_xor(x, msk, 32));
            const float mn    = fmaxf(rm[r], x);
            const float scale = __expf(rm[r] - mn);
            const float p     = __expf(pv[r] - mn);
            float ts = p;
            for (int msk = 1; msk < 16; msk <<= 1) ts += __shfl_xor(ts, msk, 32);
            rl[r] = rl[r] * scale + ts;
            rm[r] = mn;
#pragma unroll
            for (int dc = 0; dc < 4; ++dc) oacc[dc][r] *= scale;
            P32[r + hi * 8][m] = (_Float16)p;             // D-layout -> LDS
        }
        __syncthreads();

        // A = [P | 0] via zero-padded columns; V fragments contiguous from VtT
        const v16h ap = frag_a_row(&P32[m][0], 0, hi);
#pragma unroll
        for (int dc = 0; dc < 4; ++dc) {
            const v16h bv = *(const v16h*)&VtT[dc * 16 + m][hi * 16];
            oacc[dc] = wmma_f16(ap, bv, oacc[dc]);
        }
    }

#pragma unroll
    for (int r = 0; r < 8; ++r) {
        const int t = t0 + r + hi * 8;
        if (t < TT) {
            const float invl = 1.f / rl[r];
#pragma unroll
            for (int dc = 0; dc < 4; ++dc)
                o[bh + (size_t)t * CCH + dc * 16 + m] = oacc[dc][r] * invl;
        }
    }
}

// ---------------------------------------------------------------- loss
__global__ void zero1_kernel(float* p) {
    if (blockIdx.x == 0 && threadIdx.x == 0) *p = 0.f;
}

__global__ void loss_kernel(const float* __restrict__ logits, const int* __restrict__ tgt,
                            float* __restrict__ loss) {
    __shared__ float red[256];
    const int row = blockIdx.x;
    const int tid = threadIdx.x;          // V == 256 == blockDim
    const float lg = logits[(size_t)row * VV + tid];
    red[tid] = lg; __syncthreads();
    for (int o = 128; o > 0; o >>= 1) { if (tid < o) red[tid] = fmaxf(red[tid], red[tid + o]); __syncthreads(); }
    const float mx = red[0]; __syncthreads();
    red[tid] = __expf(lg - mx); __syncthreads();
    for (int o = 128; o > 0; o >>= 1) { if (tid < o) red[tid] += red[tid + o]; __syncthreads(); }
    if (tid == 0) {
        const float lse = mx + logf(red[0]);
        const float lp  = logits[(size_t)row * VV + tgt[row]] - lse;
        atomicAdd(loss, -lp / (float)MMR);
    }
}

// ---------------------------------------------------------------- launch
extern "C" void kernel_launch(void* const* d_in, const int* in_sizes, int n_in,
                              void* d_out, int out_size, void* d_ws, size_t ws_size,
                              hipStream_t stream) {
    (void)in_sizes; (void)n_in; (void)out_size; (void)ws_size;

    const int*   x     = (const int*)  d_in[0];
    const int*   tgt   = (const int*)  d_in[1];
    const float* tok   = (const float*)d_in[2];
    const float* pos   = (const float*)d_in[3];
    const float* Wq    = (const float*)d_in[4];
    const float* Wk    = (const float*)d_in[5];
    const float* Wv    = (const float*)d_in[6];
    const float* Wo    = (const float*)d_in[7];
    const float* bo    = (const float*)d_in[8];
    const float* ln1g  = (const float*)d_in[9];
    const float* ln1b  = (const float*)d_in[10];
    const float* ln2g  = (const float*)d_in[11];
    const float* ln2b  = (const float*)d_in[12];
    const float* W1    = (const float*)d_in[13];
    const float* b1    = (const float*)d_in[14];
    const float* W2    = (const float*)d_in[15];
    const float* b2    = (const float*)d_in[16];
    const float* lnfg  = (const float*)d_in[17];
    const float* lnfb  = (const float*)d_in[18];
    const float* Wf    = (const float*)d_in[19];
    const float* bf    = (const float*)d_in[20];
    float* out = (float*)d_out;

    // Workspace carve-out (256B aligned)
    char* wsp = (char*)d_ws;
    auto carve = [&](size_t bytes) -> char* {
        char* p = wsp;
        wsp += (bytes + 255) & ~(size_t)255;
        return p;
    };
    float* h  = (float*)carve((size_t)MMR * CCH * 4);
    float* z  = (float*)carve((size_t)MMR * CCH * 4);
    float* qb = (float*)carve((size_t)MMR * CCH * 4);
    float* kb = (float*)carve((size_t)MMR * CCH * 4);
    float* vb = (float*)carve((size_t)MMR * CCH * 4);
    float* ob = (float*)carve((size_t)MMR * CCH * 4);
    float* ff = (float*)carve((size_t)MMR * FFF * 4);
    _Float16* Wq16 = (_Float16*)carve((size_t)LLN * CCH * CCH * 2);   // [L][N=C][K=C]
    _Float16* Wk16 = (_Float16*)carve((size_t)LLN * CCH * CCH * 2);
    _Float16* Wv16 = (_Float16*)carve((size_t)LLN * CCH * CCH * 2);
    _Float16* Wo16 = (_Float16*)carve((size_t)LLN * CCH * CCH * 2);
    _Float16* W116 = (_Float16*)carve((size_t)LLN * CCH * FFF * 2);   // [L][N=FF][K=C]
    _Float16* W216 = (_Float16*)carve((size_t)LLN * FFF * CCH * 2);   // [L][N=C][K=FF]
    _Float16* Wf16 = (_Float16*)carve((size_t)CCH * VV * 2);          // [N=V][K=C]

    // Weight conversions with transpose (deterministic each call)
    const int ncc = LLN * CCH * CCH;
    const int ncf = LLN * CCH * FFF;
    const int nfv = CCH * VV;
    cvtT_f16_kernel<<<(ncc + 255) / 256, 256, 0, stream>>>(Wq, Wq16, CCH, CCH, LLN);
    cvtT_f16_kernel<<<(ncc + 255) / 256, 256, 0, stream>>>(Wk, Wk16, CCH, CCH, LLN);
    cvtT_f16_kernel<<<(ncc + 255) / 256, 256, 0, stream>>>(Wv, Wv16, CCH, CCH, LLN);
    cvtT_f16_kernel<<<(ncc + 255) / 256, 256, 0, stream>>>(Wo, Wo16, CCH, CCH, LLN);
    cvtT_f16_kernel<<<(ncf + 255) / 256, 256, 0, stream>>>(W1, W116, CCH, FFF, LLN);
    cvtT_f16_kernel<<<(ncf + 255) / 256, 256, 0, stream>>>(W2, W216, FFF, CCH, LLN);
    cvtT_f16_kernel<<<(nfv + 255) / 256, 256, 0, stream>>>(Wf, Wf16, CCH, VV, 1);

    embed_kernel<<<MMR, 128, 0, stream>>>(x, tok, pos, h);

    const dim3 gC(MMR / 16, CCH / 64);   // N = 384
    const dim3 gF(MMR / 16, FFF / 64);   // N = 1536
    const dim3 gV(MMR / 16, VV  / 64);   // N = 256
    const dim3 gA((TT + 15) / 16, HH, BB);

    for (int l = 0; l < LLN; ++l) {
        const size_t wcc = (size_t)l * CCH * CCH;
        ln_kernel<<<MMR, 128, 0, stream>>>(h, ln1g + l * CCH, ln1b + l * CCH, z);
        gemm_wmma_kernel<<<gC, 128, 0, stream>>>(z, Wq16 + wcc, nullptr, nullptr, qb, MMR, CCH, CCH, 0);
        gemm_wmma_kernel<<<gC, 128, 0, stream>>>(z, Wk16 + wcc, nullptr, nullptr, kb, MMR, CCH, CCH, 0);
        gemm_wmma_kernel<<<gC, 128, 0, stream>>>(z, Wv16 + wcc, nullptr, nullptr, vb, MMR, CCH, CCH, 0);
        attn_kernel<<<gA, 32, 0, stream>>>(qb, kb, vb, ob);
        gemm_wmma_kernel<<<gC, 128, 0, stream>>>(ob, Wo16 + wcc, bo + l * CCH, h, h, MMR, CCH, CCH, 0);
        ln_kernel<<<MMR, 128, 0, stream>>>(h, ln2g + l * CCH, ln2b + l * CCH, z);
        gemm_wmma_kernel<<<gF, 128, 0, stream>>>(z, W116 + (size_t)l * CCH * FFF, b1 + l * FFF,
                                                 nullptr, ff, MMR, FFF, CCH, 1);
        gemm_wmma_kernel<<<gC, 128, 0, stream>>>(ff, W216 + (size_t)l * FFF * CCH, b2 + l * CCH,
                                                 h, h, MMR, CCH, FFF, 0);
    }

    ln_kernel<<<MMR, 128, 0, stream>>>(h, lnfg, lnfb, z);
    gemm_wmma_kernel<<<gV, 128, 0, stream>>>(z, Wf16, bf, nullptr, out, MMR, VV, CCH, 0);

    zero1_kernel<<<1, 1, 0, stream>>>(out + (size_t)MMR * VV);
    loss_kernel<<<MMR, 256, 0, stream>>>(out, tgt, out + (size_t)MMR * VV);
}
